// neuron_layer_fixedpoint_8469675507797
// MI455X (gfx1250) — 8-bit matrix operand order repaired and checked by reading, compiled, not run on hardware
//
#include <hip/hip_runtime.h>
#include <stdint.h>

typedef int v8i __attribute__((ext_vector_type(8)));
typedef unsigned int u32x4 __attribute__((ext_vector_type(4)));
typedef int i32x4 __attribute__((ext_vector_type(4)));
typedef int i32x8 __attribute__((ext_vector_type(8)));

#define B_DIM 256
#define I_DIM 1024
#define N_DIM 1024
#define T_STEPS 100
#define T_OUT 99

#define WS_MAXABS 0
#define WS_VTH    4
#define WS_QB     64
#define WS_WP     8192
#define WS_SP     (2u*1024u*1024u)

#if __has_builtin(__builtin_amdgcn_tensor_load_to_lds)
#define HAVE_TDM 1
#else
#define HAVE_TDM 0
#endif

__global__ void k_init(unsigned* maxbits) {
    if (threadIdx.x == 0) *maxbits = 0u;
}

__global__ void k_maxabs(const float* __restrict__ w, unsigned* __restrict__ maxbits) {
    __shared__ unsigned red[256];
    unsigned g = blockIdx.x * 256u + threadIdx.x;
    unsigned m = 0u;
    for (unsigned i = g; i < (unsigned)(N_DIM * I_DIM); i += 65536u) {
        unsigned b = __float_as_uint(fabsf(w[i]));
        m = (b > m) ? b : m;
    }
    red[threadIdx.x] = m;
    __syncthreads();
    for (int s = 128; s > 0; s >>= 1) {
        if ((int)threadIdx.x < s) {
            unsigned o = red[threadIdx.x + s];
            if (o > red[threadIdx.x]) red[threadIdx.x] = o;
        }
        __syncthreads();
    }
    if (threadIdx.x == 0) atomicMax(maxbits, red[0]);
}

__global__ void k_quant_scalars(const float* __restrict__ bias,
                                const unsigned* __restrict__ maxbits,
                                float* __restrict__ qb, float* __restrict__ vth) {
    float scale = 255.0f / __uint_as_float(*maxbits);
    unsigned n = blockIdx.x * 256u + threadIdx.x;
    if (n < (unsigned)N_DIM) qb[n] = truncf(bias[n] * scale) * 64.0f;
    if (n == 0) *vth = truncf(scale) * 64.0f;
}

__global__ void k_pack_w(const float* __restrict__ w,
                         const unsigned* __restrict__ maxbits,
                         unsigned* __restrict__ wp) {
    float scale = 255.0f / __uint_as_float(*maxbits);
    unsigned d    = blockIdx.x * 256u + threadIdx.x;
    unsigned j    = d & 7u;
    unsigned lane = (d >> 3) & 31u;
    unsigned kc   = (d >> 8) & 15u;
    unsigned nt   = d >> 12;
    unsigned n    = nt * 16u + (lane & 15u);
    unsigned h    = lane >> 4;
    unsigned kb   = (j < 4u) ? (16u * h + 4u * j) : (32u + 16u * h + 4u * (j - 4u));
    unsigned K    = kc * 64u + kb;
    unsigned out  = 0u;
#pragma unroll
    for (int bb = 0; bb < 4; ++bb) {
        float q = truncf(w[(size_t)n * I_DIM + K + bb] * scale);
        int  k8 = (int)floorf(q * 0.5f);
        out |= ((unsigned)(k8 & 0xff)) << (8 * bb);
    }
    wp[d] = out;
}

__global__ void k_pack_s(const int* __restrict__ spk, unsigned char* __restrict__ sp) {
    __shared__ unsigned char tile[64 * T_STEPS];
    unsigned b  = blockIdx.x;
    unsigned kc = blockIdx.y;
    const int* src = spk + ((size_t)b * I_DIM + kc * 64u) * T_STEPS;
    for (unsigned l = threadIdx.x; l < 64u * T_STEPS; l += 256u)
        tile[l] = (unsigned char)src[l];
    __syncthreads();
    unsigned bt = b >> 4, bl = b & 15u;
    for (unsigned d = threadIdx.x; d < 1600u; d += 256u) {
        unsigned t   = d >> 4;
        unsigned rem = d & 15u;
        unsigned h   = rem >> 3, j = rem & 7u;
        unsigned ka  = 32u * (j >> 2) + 16u * h + 4u * (j & 3u);
        unsigned v   = 0u;
#pragma unroll
        for (int bb = 0; bb < 4; ++bb)
            v |= ((unsigned)tile[(ka + bb) * T_STEPS + t]) << (8 * bb);
        size_t off = ((((size_t)t * 16u + bt) * 16u + kc) * 32u + (h * 16u + bl)) * 8u + j;
        ((unsigned*)sp)[off] = v;
    }
}

__device__ __forceinline__ void tdm_load_tile(const unsigned char* gsrc, unsigned lds_off) {
#if HAVE_TDM
    unsigned long long ga = (unsigned long long)(uintptr_t)gsrc;
    u32x4 g0;
    g0.x = 1u;
    g0.y = lds_off;
    g0.z = (unsigned)(ga & 0xffffffffu);
    g0.w = (unsigned)((ga >> 32) & 0x01ffffffu) | (2u << 30);
    i32x8 g1;
    g1[0] = 0;
    g1[1] = (int)(16384u << 16);
    g1[2] = (int)(1u << 16);
    g1[3] = (int)(16384u << 16);
    g1[4] = 0;
    g1[5] = 16384;
    g1[6] = 0;
    g1[7] = 0;
    i32x4 gz = {0, 0, 0, 0};
#if defined(__clang_major__) && (__clang_major__ >= 23)
    i32x8 gz8 = {0, 0, 0, 0, 0, 0, 0, 0};
    __builtin_amdgcn_tensor_load_to_lds(g0, g1, gz, gz, gz8, 0);
#else
    __builtin_amdgcn_tensor_load_to_lds(g0, g1, gz, gz, 0);
#endif
#endif
}

__device__ __forceinline__ void wait_tensor0() {
#if __has_builtin(__builtin_amdgcn_s_wait_tensorcnt)
    __builtin_amdgcn_s_wait_tensorcnt(0);
#else
    asm volatile("s_wait_tensorcnt 0x0" ::: "memory");
#endif
}

__global__ void __launch_bounds__(256) k_main(const unsigned char* __restrict__ sp,
                                              const unsigned char* __restrict__ wp8,
                                              const float* __restrict__ qb,
                                              const float* __restrict__ vthp,
                                              float* __restrict__ out) {
    __shared__ struct __align__(32) {
        unsigned char abuf[2][16384];
        unsigned sbits[2048 * 4];
    } sm;

    const unsigned tid  = threadIdx.x;
    const unsigned lane = tid & 31u, wid = tid >> 5;
    const unsigned bx = blockIdx.x;
    const unsigned by = blockIdx.y;
    const unsigned h  = lane >> 4;

    const float vth  = *vthp;
    const float keep = 3897.0f / 4096.0f;
    const float VMAXc = 8388607.0f, VMINc = -8388607.0f;
    const float qb_l = qb[by * 128u + wid * 16u + (lane & 15u)];

    v8i bw[16];
    {
        const unsigned ng = by * 8u + wid;
#pragma unroll
        for (int kc = 0; kc < 16; ++kc)
            bw[kc] = *(const v8i*)(wp8 + (((size_t)ng * 16u + kc) * 32u + lane) * 32u);
    }

    float v[8];
#pragma unroll
    for (int i = 0; i < 8; ++i) v[i] = 0.0f;

    const unsigned char* tile0 = sp + (size_t)bx * 16384u;

#if HAVE_TDM
    if (wid == 0) tdm_load_tile(tile0, 0u);
    wait_tensor0();
#else
    {
        const uint4* s = (const uint4*)tile0;
        uint4* d = (uint4*)&sm.abuf[0][0];
#pragma unroll
        for (int i = 0; i < 4; ++i) d[tid + 256u * i] = s[tid + 256u * i];
    }
#endif
    __syncthreads();

    for (unsigned tw = 0; tw < 4; ++tw) {
        const unsigned tl = (tw < 3) ? 32u : 2u;
        unsigned m[8];
#pragma unroll
        for (int i = 0; i < 8; ++i) m[i] = 0u;

        for (unsigned ti = 0; ti < tl; ++ti) {
            const unsigned t = tw * 32u + ti;
            const unsigned cur = t & 1u, nxt = cur ^ 1u;

#if HAVE_TDM
            if (t < 97u && wid == 0)
                tdm_load_tile(sp + ((size_t)(t + 1) * 16u + bx) * 16384u,
                              nxt * 16384u);
#endif
            const unsigned char* abase = &sm.abuf[cur][lane * 32u];

            v8i acc0 = {0, 0, 0, 0, 0, 0, 0, 0};
            v8i acc1 = {0, 0, 0, 0, 0, 0, 0, 0};
            v8i a0 = *(const v8i*)(abase);
#pragma unroll
            for (int kc = 0; kc < 16; kc += 2) {
                v8i a1 = *(const v8i*)(abase + (size_t)(kc + 1) * 1024u);
                acc0 = __builtin_amdgcn_wmma_i32_16x16x64_iu8(
                           false, a0, true, bw[kc], acc0, false, false);
                if (kc + 2 < 16)
                    a0 = *(const v8i*)(abase + (size_t)(kc + 2) * 1024u);
                acc1 = __builtin_amdgcn_wmma_i32_16x16x64_iu8(
                           false, a1, true, bw[kc + 1], acc1, false, false);
            }
            v8i acc = acc0 + acc1;

#pragma unroll
            for (int i = 0; i < 8; ++i) {
                float wi = 128.0f * (float)acc[i];
                float nv = v[i] * keep + wi + qb_l;
                nv = fminf(fmaxf(nv, VMINc), VMAXc);
                unsigned spkbit = (nv >= vth) ? 1u : 0u;
                m[i] |= spkbit << ti;
                v[i] = (nv > vth) ? (nv - vth) : nv;
            }

            if (t < 97u) {
#if HAVE_TDM
                wait_tensor0();
                __syncthreads();
#else
                __syncthreads();
                const uint4* s2 = (const uint4*)(sp + ((size_t)(t + 1) * 16u + bx) * 16384u);
                uint4* d2 = (uint4*)&sm.abuf[nxt][0];
#pragma unroll
                for (int i = 0; i < 4; ++i) d2[tid + 256u * i] = s2[tid + 256u * i];
                __syncthreads();
#endif
            }
        }
#pragma unroll
        for (int i = 0; i < 8; ++i) {
            unsigned bloc = (unsigned)i + 8u * h;
            unsigned nloc = wid * 16u + (lane & 15u);
            sm.sbits[(bloc * 128u + nloc) * 4u + tw] = m[i];
        }
    }
    __syncthreads();

    for (unsigned k = 0; k < 256; ++k) {
        unsigned p  = wid * 256u + k;
        unsigned w0 = sm.sbits[p * 4 + 0], w1 = sm.sbits[p * 4 + 1];
        unsigned w2 = sm.sbits[p * 4 + 2], w3 = sm.sbits[p * 4 + 3];
        size_t base = (((size_t)(bx * 16u + (p >> 7))) * N_DIM
                       + by * 128u + (p & 127u)) * T_OUT;
        if (lane == 0) out[base] = 0.0f;
        out[base + 1u  + lane] = (float)((w0 >> lane) & 1u);
        out[base + 33u + lane] = (float)((w1 >> lane) & 1u);
        out[base + 65u + lane] = (float)((w2 >> lane) & 1u);
        if (lane < 2u) out[base + 97u + lane] = (float)((w3 >> lane) & 1u);
    }
}

extern "C" void kernel_launch(void* const* d_in, const int* in_sizes, int n_in,
                              void* d_out, int out_size, void* d_ws, size_t ws_size,
                              hipStream_t stream) {
    (void)in_sizes; (void)n_in; (void)out_size; (void)ws_size;
    const int*   spikes = (const int*)d_in[0];
    const float* weight = (const float*)d_in[1];
    const float* bias   = (const float*)d_in[2];
    float* out = (float*)d_out;

    char* ws = (char*)d_ws;
    unsigned*      maxbits = (unsigned*)(ws + WS_MAXABS);
    float*         vth     = (float*)(ws + WS_VTH);
    float*         qb      = (float*)(ws + WS_QB);
    unsigned char* wp8     = (unsigned char*)(ws + WS_WP);
    unsigned char* spb     = (unsigned char*)(ws + WS_SP);

    k_init<<<dim3(1), dim3(1), 0, stream>>>(maxbits);
    k_maxabs<<<dim3(256), dim3(256), 0, stream>>>(weight, maxbits);
    k_quant_scalars<<<dim3(4), dim3(256), 0, stream>>>(bias, maxbits, qb, vth);
    k_pack_w<<<dim3(1024), dim3(256), 0, stream>>>(weight, maxbits, (unsigned*)wp8);
    k_pack_s<<<dim3(256, 16), dim3(256), 0, stream>>>(spikes, spb);
    k_main<<<dim3(16, 8), dim3(256), 0, stream>>>(spb, wp8, qb, vth, out);
}
